// BiLSTMClassifier_57320633533002
// MI455X (gfx1250) — compile-verified, matrix operand order checked by reading, not run on hardware
//
#include <hip/hip_runtime.h>
#include <stdint.h>

// ---------------- problem constants ----------------
static constexpr int kB = 64;     // batch
static constexpr int kT = 1024;   // seq len
static constexpr int kD = 256;    // input dim
static constexpr int kH = 256;    // hidden dim per direction
static constexpr int kL = 2;      // labels

#define GAS __attribute__((address_space(1)))

// ---------------- WMMA plumbing ----------------
typedef __attribute__((ext_vector_type(16))) __bf16    v16bf;
typedef __attribute__((ext_vector_type(8)))  float     v8f;
typedef __attribute__((ext_vector_type(4)))  uint32_t  v4u;

union AFrag {
  v16bf v;
  v4u   q2[2];
  uint32_t u[8];
};

__device__ __forceinline__ v8f wmma_bf16(const AFrag& a, const AFrag& b, v8f c) {
  return __builtin_amdgcn_wmma_f32_16x16x32_bf16(
      false, a.v, false, b.v, (short)0, c, false, false);
}

// Fragment-packed layout: one 16x32 tile = one "block" of 32 lanes x 8 dwords (1KB),
// dword index = blk*256 + lane*8 + p; lane = hi*16 + rowInTile; dword p holds the
// bf16 pair at kk = hi*8 + 2*(p&3) + (p>=4 ? 16 : 0), k = blk_col*32 + kk.
__device__ __forceinline__ AFrag ldp(const uint32_t* p) {
  AFrag f;
  const GAS v4u* q = (const GAS v4u*)p;
  f.q2[0] = q[0];
  f.q2[1] = q[1];
  return f;
}

// ushort index of element (rowInTile, colGlobal) inside a fragment-packed matrix.
__device__ __forceinline__ size_t packed_idx(size_t blkBase, int rowInTile, int colGlobal) {
  const int s    = colGlobal >> 5;
  const int half = colGlobal & 1;
  const int e    = colGlobal & 30;
  const int hi   = (e >> 3) & 1;
  const int pp   = ((e & 7) >> 1) + ((e >= 16) ? 4 : 0);
  const int lane = hi * 16 + rowInTile;
  return ((blkBase + (size_t)s) * 256 + (size_t)lane * 8 + pp) * 2 + half;
}

__device__ __forceinline__ void store_g16(unsigned short* p, unsigned short v) {
  *(GAS unsigned short*)p = v;
}
__device__ __forceinline__ void store_g32(uint32_t* p, uint32_t v) {
  *(GAS uint32_t*)p = v;
}

// ---------------- bf16 helpers ----------------
__device__ __forceinline__ unsigned short f2bf(float f) {
  union { float f; uint32_t u; } v; v.f = f;
  uint32_t r = v.u + 0x7FFFu + ((v.u >> 16) & 1u);
  return (unsigned short)(r >> 16);
}
__device__ __forceinline__ float bf2f(unsigned short h) {
  union { uint32_t u; float f; } v; v.u = ((uint32_t)h) << 16;
  return v.f;
}
__device__ __forceinline__ float fsigmoid(float x) {
  return 1.0f / (1.0f + __expf(-x));
}

// ---------------- multi-workgroup barrier (sense-reversal through L2) ----------------
__device__ __forceinline__ void grid_barrier(unsigned* cnt, unsigned* gen, unsigned nb) {
  __threadfence();
  __syncthreads();
  if (threadIdx.x == 0) {
    const unsigned g = __hip_atomic_load(gen, __ATOMIC_RELAXED, __HIP_MEMORY_SCOPE_AGENT);
    const unsigned a = __hip_atomic_fetch_add(cnt, 1u, __ATOMIC_ACQ_REL, __HIP_MEMORY_SCOPE_AGENT);
    if (a == nb - 1u) {
      __hip_atomic_store(cnt, 0u, __ATOMIC_RELAXED, __HIP_MEMORY_SCOPE_AGENT);
      __hip_atomic_fetch_add(gen, 1u, __ATOMIC_RELEASE, __HIP_MEMORY_SCOPE_AGENT);
    } else {
      while (__hip_atomic_load(gen, __ATOMIC_ACQUIRE, __HIP_MEMORY_SCOPE_AGENT) == g)
        __builtin_amdgcn_s_sleep(1);
    }
  }
  __syncthreads();
}

// ---------------- multi-block persistent LSTM scan ----------------
// NB blocks x 8 waves; one 16x16 (batch,hidden) unit per wave; the wave owns the i,f,g,o
// gate tiles of its unit; c stays in 8 VGPRs across all T steps. Hidden state is
// double-buffered in fragment-packed global buffers (L2); ONE grid barrier per step.
template <int HC, int DIN, int NB>
__device__ __forceinline__ void lstm_scan_mb(
    const uint32_t* __restrict__ xpack,  // packed A: [T][4][SX] blocks
    const uint32_t* __restrict__ wpack,  // packed B: [HT][KT][4] blocks (Wi steps then Wh)
    const float*    __restrict__ bias,   // f32 [4*HC]
    uint32_t* __restrict__ h0,           // packed A: [4][SH] blocks
    uint32_t* __restrict__ h1,
    unsigned short* __restrict__ out,    // storeAll: packed comb; else plain [64][512] hlast
    int outColOff, int storeAll, int rev, int T,
    unsigned* cnt, unsigned* gen, int bid) {
  constexpr int HT = HC / 16;
  constexpr int SX = DIN / 32;
  constexpr int SH = HC / 32;
  constexpr int KT = SX + SH;
  static_assert(4 * HT == NB * 8, "one unit per wave");
  static_assert((KT & 1) == 0, "2-stage pipeline needs even KT");

  const int tid   = threadIdx.x;
  const int lane  = tid & 31;
  const int wave  = tid >> 5;
  const int lmod  = lane & 15;
  const int rbase = (lane >> 4) * 8;

  const int unit = bid * 8 + wave;
  const int mt   = unit / HT;
  const int ht   = unit % HT;
  const int m0   = mt * 16;
  const int ncol = ht * 16 + lmod;

  // zero h_0 (packed), cooperatively across the group
  for (int i = bid * 256 + tid; i < 4 * SH * 256; i += NB * 256)
    store_g32(h0 + i, 0u);

  float bv[4];
#pragma unroll
  for (int q = 0; q < 4; ++q) bv[q] = bias[q * HC + ncol];

  float c[8];
#pragma unroll
  for (int r = 0; r < 8; ++r) c[r] = 0.0f;

  // t-invariant per-lane base pointers
  const uint32_t* wp  = wpack + ((size_t)ht * KT * 4) * 256 + (size_t)lane * 8;
  const uint32_t* hp0 = h0 + (size_t)mt * SH * 256 + (size_t)lane * 8;
  const uint32_t* hp1 = h1 + (size_t)mt * SH * 256 + (size_t)lane * 8;
  const size_t hblk = (size_t)mt * SH;

  grid_barrier(cnt, gen, NB);   // h_0 visible everywhere

  for (int t = 0; t < T; ++t) {
    const int tread = rev ? (T - 1 - t) : t;
    const uint32_t* hp = (t & 1) ? hp1 : hp0;
    uint32_t*    hnext = (t & 1) ? h0 : h1;

    const uint32_t* xp =
        xpack + (((size_t)tread * 4 + mt) * SX) * 256 + (size_t)lane * 8;

    if (t + 1 < T) {  // prefetch next timestep's x fragment block
      const int tn = rev ? (T - 2 - t) : (t + 1);
      __builtin_prefetch(
          (const void*)(xpack + (((size_t)tn * 4 + mt) * SX) * 256 + (size_t)lane * 8), 0, 3);
    }

    auto loadA = [&](int s) -> AFrag {
      return (s < SX) ? ldp(xp + (size_t)s * 256) : ldp(hp + (size_t)(s - SX) * 256);
    };

    v8f acc[4];
#pragma unroll
    for (int q = 0; q < 4; ++q)
#pragma unroll
      for (int r = 0; r < 8; ++r) acc[q][r] = bv[q];   // bias fused into C init

    // ---- 2-stage modulo-scheduled K loop: no fragment rotation copies ----
    const uint32_t* wq = wp;
    AFrag a0  = loadA(0);
    AFrag b00 = ldp(wq + 0);
    AFrag b01 = ldp(wq + 256);
    AFrag b02 = ldp(wq + 512);
    AFrag b03 = ldp(wq + 768);

#pragma unroll 1
    for (int s = 0; s < KT; s += 2) {
      // stage-1 fragments (s+1 < KT always: KT even)
      AFrag a1  = loadA(s + 1);
      AFrag b10 = ldp(wq + 1024);
      AFrag b11 = ldp(wq + 1280);
      AFrag b12 = ldp(wq + 1536);
      AFrag b13 = ldp(wq + 1792);

      acc[0] = wmma_bf16(a0, b00, acc[0]);
      acc[1] = wmma_bf16(a0, b01, acc[1]);
      acc[2] = wmma_bf16(a0, b02, acc[2]);
      acc[3] = wmma_bf16(a0, b03, acc[3]);

      // stage-0 lookahead for s+2 (branch-free clamp to block 0 on last iteration)
      const bool more = (s + 2) < KT;
      const uint32_t* wn = more ? (wq + 2048) : wp;
      a0  = loadA(more ? (s + 2) : 0);
      b00 = ldp(wn + 0);
      b01 = ldp(wn + 256);
      b02 = ldp(wn + 512);
      b03 = ldp(wn + 768);

      acc[0] = wmma_bf16(a1, b10, acc[0]);
      acc[1] = wmma_bf16(a1, b11, acc[1]);
      acc[2] = wmma_bf16(a1, b12, acc[2]);
      acc[3] = wmma_bf16(a1, b13, acc[3]);

      wq = wn;
    }

    // ---- register-only cell update (gate order i,f,g,o) ----
    float hv[8];
#pragma unroll
    for (int r = 0; r < 8; ++r) {
      const float iv = fsigmoid(acc[0][r]);
      const float fv = fsigmoid(acc[1][r]);
      const float gv = tanhf(acc[2][r]);
      const float ov = fsigmoid(acc[3][r]);
      const float cc = fv * c[r] + iv * gv;
      c[r]  = cc;
      hv[r] = ov * tanhf(cc);
    }

#pragma unroll
    for (int r = 0; r < 8; ++r) {
      const int rit = rbase + r;                // row inside the 16x16 tile
      const unsigned short hb = f2bf(hv[r]);
      store_g16((unsigned short*)hnext + packed_idx(hblk, rit, ncol), hb);
      if (storeAll) {
        store_g16(out + packed_idx(((size_t)t * 4 + mt) * 16, rit, outColOff + ncol), hb);
      } else if (t == T - 1) {
        store_g16(out + (size_t)(m0 + rit) * (2 * kH) + outColOff + ncol, hb);
      }
    }

    grid_barrier(cnt, gen, NB);   // publish h_t; protect buffer reused at t+1
  }
}

// ---------------- kernels ----------------
__global__ void __launch_bounds__(256, 1)
bilstm_dir_kernel(const uint32_t* __restrict__ xpack,
                  const uint32_t* __restrict__ wpf, const float* __restrict__ bf,
                  const uint32_t* __restrict__ wpb, const float* __restrict__ bb,
                  uint32_t* __restrict__ hf0, uint32_t* __restrict__ hf1,
                  uint32_t* __restrict__ hr0, uint32_t* __restrict__ hr1,
                  unsigned short* __restrict__ comb,
                  unsigned* __restrict__ flags, int T) {
  const int gb = blockIdx.x;
  if (gb < 8) {  // forward direction -> comb columns [0,256)
    lstm_scan_mb<kH, kD, 8>(xpack, wpf, bf, hf0, hf1, comb,
                            0, 1, 0, T, flags + 0, flags + 1, gb);
  } else {       // backward direction (scan order == reference pairing) -> cols [256,512)
    lstm_scan_mb<kH, kD, 8>(xpack, wpb, bb, hr0, hr1, comb,
                            kH, 1, 1, T, flags + 64, flags + 65, gb - 8);
  }
}

__global__ void __launch_bounds__(256, 1)
bilstm_comb_kernel(const uint32_t* __restrict__ combp,
                   const uint32_t* __restrict__ wps, const float* __restrict__ bs,
                   uint32_t* __restrict__ hc0, uint32_t* __restrict__ hc1,
                   unsigned short* __restrict__ hlast,
                   unsigned* __restrict__ flags, int T) {
  lstm_scan_mb<2 * kH, 2 * kH, 16>(combp, wps, bs, hc0, hc1, hlast,
                                   0, 0, 0, T, flags + 128, flags + 129, (int)blockIdx.x);
}

__global__ void init_flags_kernel(unsigned* __restrict__ flags, int n) {
  const int i = blockIdx.x * blockDim.x + threadIdx.x;
  if (i < n) flags[i] = 0;
}

// Pack fp32 Wi[4*HC][DIN] ‖ Wh[4*HC][HC] into bf16 fragment blocks [ht][s][q][lane][8dw].
__global__ void pack_w_kernel(const float* __restrict__ Wi, const float* __restrict__ Wh,
                              uint32_t* __restrict__ dst, int HC, int DIN, int nDwords) {
  const int idx = blockIdx.x * blockDim.x + threadIdx.x;
  if (idx >= nDwords) return;
  const int SX = DIN / 32, KT = SX + HC / 32;
  const int p    = idx & 7;
  const int lane = (idx >> 3) & 31;
  const int q    = (idx >> 8) & 3;
  const int rest = idx >> 10;
  const int s    = rest % KT;
  const int ht   = rest / KT;
  const int row  = q * HC + ht * 16 + (lane & 15);
  const int kk   = ((lane >> 4) << 3) + ((p & 3) << 1) + ((p >= 4) ? 16 : 0);
  const float* src = (s < SX) ? (Wi + (size_t)row * DIN + s * 32 + kk)
                              : (Wh + (size_t)row * HC + (s - SX) * 32 + kk);
  store_g32(dst + idx, (uint32_t)f2bf(src[0]) | ((uint32_t)f2bf(src[1]) << 16));
}

// Pack x [B][T][D] fp32 into bf16 fragment blocks [t][mt][sx][lane][8dw].
__global__ void pack_x_kernel(const float* __restrict__ x, uint32_t* __restrict__ dst, int nDwords) {
  const int idx = blockIdx.x * blockDim.x + threadIdx.x;
  if (idx >= nDwords) return;
  const int p    = idx & 7;
  const int lane = (idx >> 3) & 31;
  const int sx   = (idx >> 8) & 7;    // SX = 8
  const int mt   = (idx >> 11) & 3;
  const int t    = idx >> 13;
  const int b    = mt * 16 + (lane & 15);
  const int k    = sx * 32 + ((lane >> 4) << 3) + ((p & 3) << 1) + ((p >= 4) ? 16 : 0);
  const float* src = x + ((size_t)b * kT + t) * kD + k;
  store_g32(dst + idx, (uint32_t)f2bf(src[0]) | ((uint32_t)f2bf(src[1]) << 16));
}

__global__ void head_kernel(const unsigned short* __restrict__ hlast,
                            const float* __restrict__ Wo, const float* __restrict__ bo,
                            float* __restrict__ out) {
  const int tid = threadIdx.x;
  if (tid >= kB * kL) return;
  const int b = tid >> 1;
  const int l = tid & 1;
  float s = bo[l];
  for (int k = 0; k < 2 * kH; ++k)
    s += bf2f(hlast[b * (2 * kH) + k]) * Wo[l * (2 * kH) + k];
  out[b * kL + l] = 1.0f / (1.0f + __expf(-s));
}

// ---------------- host launcher ----------------
extern "C" void kernel_launch(void* const* d_in, const int* in_sizes, int n_in,
                              void* d_out, int out_size, void* d_ws, size_t ws_size,
                              hipStream_t stream) {
  (void)in_sizes; (void)n_in; (void)out_size; (void)ws_size;

  const float* x   = (const float*)d_in[0];
  const float* Wif = (const float*)d_in[1];
  const float* Whf = (const float*)d_in[2];
  const float* bf  = (const float*)d_in[3];
  const float* Wib = (const float*)d_in[4];
  const float* Whb = (const float*)d_in[5];
  const float* bb  = (const float*)d_in[6];
  const float* Wis = (const float*)d_in[7];
  const float* Whs = (const float*)d_in[8];
  const float* bs  = (const float*)d_in[9];
  const float* Wo  = (const float*)d_in[10];
  const float* bo  = (const float*)d_in[11];

  unsigned char* ws = (unsigned char*)d_ws;
  size_t off = 0;
  auto carve = [&](size_t bytes) -> void* {
    void* p = ws + off;
    off += (bytes + 255) & ~(size_t)255;
    return p;
  };

  // packed sizes (in dwords)
  const int nXp  = kT * 4 * (kD / 32) * 256;                 // 8,388,608 dw (32 MB)
  const int nWpD = (kH / 16) * ((kD + kH) / 32) * 4 * 256;   //   262,144 dw (1 MB) per dir
  const int nWpS = (2 * kH / 16) * ((4 * kH) / 32) * 4 * 256;// 1,048,576 dw (4 MB)
  const int nCp  = kT * 4 * 16 * 256;                        // 16,777,216 dw (64 MB)
  const int nHd  = 4 * (kH / 32) * 256;                      // 8,192 dw per buffer
  const int nHc  = 4 * (2 * kH / 32) * 256;                  // 16,384 dw per buffer

  unsigned* flags = (unsigned*)carve(256 * sizeof(unsigned));
  uint32_t* xpack = (uint32_t*)carve((size_t)nXp * 4);
  uint32_t* combp = (uint32_t*)carve((size_t)nCp * 4);
  uint32_t* wpf   = (uint32_t*)carve((size_t)nWpD * 4);
  uint32_t* wpb   = (uint32_t*)carve((size_t)nWpD * 4);
  uint32_t* wps   = (uint32_t*)carve((size_t)nWpS * 4);
  uint32_t* hf0   = (uint32_t*)carve((size_t)nHd * 4);
  uint32_t* hf1   = (uint32_t*)carve((size_t)nHd * 4);
  uint32_t* hr0   = (uint32_t*)carve((size_t)nHd * 4);
  uint32_t* hr1   = (uint32_t*)carve((size_t)nHd * 4);
  uint32_t* hc0   = (uint32_t*)carve((size_t)nHc * 4);
  uint32_t* hc1   = (uint32_t*)carve((size_t)nHc * 4);
  unsigned short* hlast = (unsigned short*)carve((size_t)kB * 2 * kH * 2);

  const int CB = 256;
  init_flags_kernel<<<1, CB, 0, stream>>>(flags, 256);

  pack_w_kernel<<<(nWpD + CB - 1) / CB, CB, 0, stream>>>(Wif, Whf, wpf, kH, kD, nWpD);
  pack_w_kernel<<<(nWpD + CB - 1) / CB, CB, 0, stream>>>(Wib, Whb, wpb, kH, kD, nWpD);
  pack_w_kernel<<<(nWpS + CB - 1) / CB, CB, 0, stream>>>(Wis, Whs, wps, 2 * kH, 2 * kH, nWpS);
  pack_x_kernel<<<(nXp + CB - 1) / CB, CB, 0, stream>>>(x, xpack, nXp);

  // fwd (blocks 0-7) + bwd (blocks 8-15), independent barrier slots
  bilstm_dir_kernel<<<16, 256, 0, stream>>>(
      xpack, wpf, bf, wpb, bb, hf0, hf1, hr0, hr1,
      (unsigned short*)combp, flags, kT);

  // combiner scan across 16 workgroups
  bilstm_comb_kernel<<<16, 256, 0, stream>>>(
      combp, wps, bs, hc0, hc1, hlast, flags, kT);

  head_kernel<<<1, 128, 0, stream>>>(hlast, Wo, bo, (float*)d_out);
}
